// FocDecoderRNN_23733989278171
// MI455X (gfx1250) — matrix operand order repaired and checked by reading, not compiled, not run on hardware
//
#include <hip/hip_runtime.h>
#include <hip/hip_bf16.h>
#include <stdint.h>

// ---------------- CDNA5 WMMA types ----------------
typedef __attribute__((ext_vector_type(16))) __bf16 v16bf;
typedef __attribute__((ext_vector_type(8)))  float  v8f;

#define B_    2048
#define H_    256
#define GRID_ 8
#define S_    8

// LDS row = 72 bf16 (36 dwords): 64 used + pad. 16B-aligned b128 ops,
// 36*r mod 64 distinct for r=0..15 -> conflict-free column reads.
#define LDSW 36

__device__ __forceinline__ unsigned pack_bf16x2(float a, float b) {
    unsigned ua = __float_as_uint(a);
    unsigned ub = __float_as_uint(b);
    ua = (ua + 0x7FFFu + ((ua >> 16) & 1u)) >> 16;   // RNE f32->bf16
    ub = (ub + 0x7FFFu + ((ub >> 16) & 1u)) >> 16;
    return (ua & 0xFFFFu) | (ub << 16);
}

union FragB16 { uint4 q[2]; v16bf v; };

// C[M][ldc] = A[M][lda] @ W[N][ldw]^T + bias[N]
// Block tile 128x64, BK=64 per barrier pair. 256 threads = 8 waves (4M x 2N),
// each wave computes a 32x32 output (4 WMMA accumulators, 8 WMMA/iter).
// Requires M%128==0, N%64==0, K%64==0 (all call sites satisfy this).
__global__ __launch_bounds__(256)
void gemm_bf16_wmma(const float* __restrict__ A, int lda,
                    const float* __restrict__ W, int ldw,
                    const float* __restrict__ bias,
                    float* __restrict__ C, int ldc, int K)
{
    __shared__ unsigned lA[128 * LDSW];
    __shared__ unsigned lB[64 * LDSW];

    const int t    = threadIdx.x;
    const int lane = t & 31;
    const int w    = t >> 5;
    const int wm   = w & 3;    // 32-row sub-tile
    const int wn   = w >> 2;   // 32-col sub-tile
    const int m0   = blockIdx.y * 128;
    const int n0   = blockIdx.x * 64;

    const int l15  = lane & 15;
    const int half = lane >> 4;

    v8f acc00 = {}, acc01 = {}, acc10 = {}, acc11 = {};

    const unsigned* arow0 = &lA[(wm * 32 +      l15) * LDSW];
    const unsigned* arow1 = &lA[(wm * 32 + 16 + l15) * LDSW];
    const unsigned* brow0 = &lB[(wn * 32 +      l15) * LDSW];
    const unsigned* brow1 = &lB[(wn * 32 + 16 + l15) * LDSW];

    for (int k0 = 0; k0 < K; k0 += 64) {
        // ---- cooperative global->LDS load with f32->bf16 convert ----
        // A: 128 rows x 64 k = 1024 8-float segments; B: 64 rows = 512 segs.
#pragma unroll
        for (int i = 0; i < 4; ++i) {
            int s   = t + 256 * i;
            int row = s >> 3;
            int col = (s & 7) * 8;
            const float* p = A + (size_t)(m0 + row) * lda + k0 + col;
            float4 f0 = ((const float4*)p)[0];
            float4 f1 = ((const float4*)p)[1];
            uint4 q = { pack_bf16x2(f0.x, f0.y), pack_bf16x2(f0.z, f0.w),
                        pack_bf16x2(f1.x, f1.y), pack_bf16x2(f1.z, f1.w) };
            *(uint4*)&lA[row * LDSW + (col >> 1)] = q;
        }
#pragma unroll
        for (int i = 0; i < 2; ++i) {
            int s   = t + 256 * i;
            int row = s >> 3;
            int col = (s & 7) * 8;
            const float* p = W + (size_t)(n0 + row) * ldw + k0 + col;
            float4 f0 = ((const float4*)p)[0];
            float4 f1 = ((const float4*)p)[1];
            uint4 q = { pack_bf16x2(f0.x, f0.y), pack_bf16x2(f0.z, f0.w),
                        pack_bf16x2(f1.x, f1.y), pack_bf16x2(f1.z, f1.w) };
            *(uint4*)&lB[row * LDSW + (col >> 1)] = q;
        }
        __syncthreads();

#pragma unroll
        for (int kh = 0; kh < 2; ++kh) {
            const int kofs = kh * 16;   // dwords per 32-K sub-tile
            // A fragments (16x32 bf16): reg v -> dword (v&3)+8*(v>>2)+4*half
            FragB16 a0, a1, b0, b1;
            a0.q[0] = *(const uint4*)(arow0 + kofs + 4 * half);
            a0.q[1] = *(const uint4*)(arow0 + kofs + 8 + 4 * half);
            a1.q[0] = *(const uint4*)(arow1 + kofs + 4 * half);
            a1.q[1] = *(const uint4*)(arow1 + kofs + 8 + 4 * half);
            // B fragments (32x16 bf16): lane=col, reg v -> dword (v&3)+8*(v>>2)+4*half
            b0.q[0] = *(const uint4*)(brow0 + kofs + 4 * half);
            b0.q[1] = *(const uint4*)(brow0 + kofs + 8 + 4 * half);
            b1.q[0] = *(const uint4*)(brow1 + kofs + 4 * half);
            b1.q[1] = *(const uint4*)(brow1 + kofs + 8 + 4 * half);

            acc00 = __builtin_amdgcn_wmma_f32_16x16x32_bf16(false, a0.v, false, b0.v,
                                                            (short)0, acc00, false, false);
            acc01 = __builtin_amdgcn_wmma_f32_16x16x32_bf16(false, a0.v, false, b1.v,
                                                            (short)0, acc01, false, false);
            acc10 = __builtin_amdgcn_wmma_f32_16x16x32_bf16(false, a1.v, false, b0.v,
                                                            (short)0, acc10, false, false);
            acc11 = __builtin_amdgcn_wmma_f32_16x16x32_bf16(false, a1.v, false, b1.v,
                                                            (short)0, acc11, false, false);
        }
        __syncthreads();
    }

    // ---- store: lane l -> col l&15; reg r -> row r + 8*(l>>4) ----
    const int nc0 = n0 + wn * 32 + l15;
    const int nc1 = nc0 + 16;
    const float bv0 = bias[nc0];
    const float bv1 = bias[nc1];
    const int mb0 = m0 + wm * 32 + 8 * half;
    const int mb1 = mb0 + 16;
#pragma unroll
    for (int r = 0; r < 8; ++r) {
        C[(size_t)(mb0 + r) * ldc + nc0] = acc00[r] + bv0;
        C[(size_t)(mb0 + r) * ldc + nc1] = acc01[r] + bv1;
        C[(size_t)(mb1 + r) * ldc + nc0] = acc10[r] + bv0;
        C[(size_t)(mb1 + r) * ldc + nc1] = acc11[r] + bv1;
    }
}

// GRU pointwise: hout[m][j], gi[M][768], gh[Mh][768], h0[Mh][256], mh = m % Mh
__global__ __launch_bounds__(256)
void gru_ew(const float* __restrict__ gi, const float* __restrict__ gh,
            const float* __restrict__ h0, float* __restrict__ hout,
            int M, int Mh)
{
    int idx = blockIdx.x * blockDim.x + threadIdx.x;
    if (idx >= M * H_) return;
    int m  = idx >> 8;
    int j  = idx & 255;
    int mh = m % Mh;
    size_t gb = (size_t)m * 768 + j;
    size_t hb = (size_t)mh * 768 + j;
    float ir = gi[gb], iz = gi[gb + 256], in_ = gi[gb + 512];
    float hr = gh[hb], hz = gh[hb + 256], hn  = gh[hb + 512];
    float r = 1.f / (1.f + expf(-(ir + hr)));
    float z = 1.f / (1.f + expf(-(iz + hz)));
    float n = tanhf(in_ + r * hn);
    hout[idx] = (1.f - z) * n + z * h0[(size_t)mh * 256 + j];
}

// cells list in np.argwhere row-major order: cells[0..63]=cx, cells[64..127]=cy
__global__ void cells_kernel(const float* __restrict__ focal, int* __restrict__ cells)
{
    if (threadIdx.x == 0 && blockIdx.x == 0) {
        int k = 0;
        for (int cx = 0; cx < GRID_; ++cx)
            for (int cy = 0; cy < GRID_; ++cy)
                if (focal[cx * GRID_ + cy] > 0.5f) { cells[k] = cx; cells[64 + k] = cy; ++k; }
    }
}

// y[b][i][j] = o1[b][i]
__global__ __launch_bounds__(256)
void bcast_y(const float* __restrict__ o1, float* __restrict__ y)
{
    int idx = blockIdx.x * blockDim.x + threadIdx.x;   // B*4096
    int b = idx >> 12;
    int i = (idx & 4095) >> 6;
    y[idx] = o1[b * 64 + i];
}

// patches[kc][b][si*8+sj] = x[b][ (cx*8+si)*64 + cy*8+sj ]
__global__ __launch_bounds__(256)
void gather_patches(const float* __restrict__ x, const int* __restrict__ cells,
                    int chunkBase, float* __restrict__ patches, int total)
{
    int idx = blockIdx.x * blockDim.x + threadIdx.x;
    if (idx >= total) return;
    int s  = idx & 63;
    int b  = (idx >> 6) & (B_ - 1);
    int kc = idx >> 17;
    int kk = chunkBase + kc;
    int cx = cells[kk], cy = cells[64 + kk];
    int si = s >> 3, sj = s & 7;
    patches[idx] = x[(size_t)b * 4096 + (cx * 8 + si) * 64 + cy * 8 + sj];
}

// y[b][(cx*8+si)*64 + cy*8+sj] += o2[idx]   (cells disjoint -> no atomics)
__global__ __launch_bounds__(256)
void scatter_o2(const float* __restrict__ o2, const int* __restrict__ cells,
                int chunkBase, float* __restrict__ y, int total)
{
    int idx = blockIdx.x * blockDim.x + threadIdx.x;
    if (idx >= total) return;
    int s  = idx & 63;
    int b  = (idx >> 6) & (B_ - 1);
    int kc = idx >> 17;
    int kk = chunkBase + kc;
    int cx = cells[kk], cy = cells[64 + kk];
    int si = s >> 3, sj = s & 7;
    y[(size_t)b * 4096 + (cx * 8 + si) * 64 + cy * 8 + sj] += o2[idx];
}

extern "C" void kernel_launch(void* const* d_in, const int* in_sizes, int n_in,
                              void* d_out, int out_size, void* d_ws, size_t ws_size,
                              hipStream_t stream)
{
    const float* x        = (const float*)d_in[0];
    const float* h        = (const float*)d_in[1];   // (1,B,H) flat = B*256
    const float* focal    = (const float*)d_in[2];
    const float* embed1_w = (const float*)d_in[3];
    const float* embed1_b = (const float*)d_in[4];
    const float* w_ih1    = (const float*)d_in[5];
    const float* w_hh1    = (const float*)d_in[6];
    const float* b_ih1    = (const float*)d_in[7];
    const float* b_hh1    = (const float*)d_in[8];
    const float* out1_w   = (const float*)d_in[9];
    const float* out1_b   = (const float*)d_in[10];
    const float* embed2_w = (const float*)d_in[11];
    const float* embed2_b = (const float*)d_in[12];
    const float* w_ih2    = (const float*)d_in[13];
    const float* w_hh2    = (const float*)d_in[14];
    const float* b_ih2    = (const float*)d_in[15];
    const float* b_hh2    = (const float*)d_in[16];
    const float* out2_w   = (const float*)d_in[17];
    const float* out2_b   = (const float*)d_in[18];

    float* y  = (float*)d_out;                           // B*4096
    float* h1 = y + (size_t)B_ * 4096;                   // B*256  (y_h[0])
    float* h2 = h1 + (size_t)B_ * 256;                   // K*B*256

    // K recoverable from out_size: out = B*4096 + (1+K)*B*256
    int K = (int)(((long long)out_size - (long long)B_ * 4096) / ((long long)B_ * 256)) - 1;
    if (K < 0) K = 0;
    if (K > 64) K = 64;

    // ---- workspace carve-up (cells ints in first 1KB) ----
    int*   cells = (int*)d_ws;
    float* fws   = (float*)d_ws + 256;

    const size_t fixedF = (size_t)B_ * 256 + (size_t)B_ * 768 + (size_t)B_ * 64;
    const long long perCH = (long long)B_ * 1152;  // gi(768)+patches(64)+e2(256)+o2(64)
    long long availF = (long long)(ws_size / 4) - 256 - (long long)fixedF;
    int CH = 1;
    if (availF > perCH) {
        long long c = availF / perCH;
        if (c > 8) c = 8;
        if (c < 1) c = 1;
        CH = (int)c;
    }

    float* x1      = fws;
    float* gh      = x1 + (size_t)B_ * 256;              // gh1, then gh2
    float* o1      = gh + (size_t)B_ * 768;
    float* gi      = o1 + (size_t)B_ * 64;               // CH*B*768 (>= B*768)
    float* patches = gi + (size_t)CH * B_ * 768;         // CH*B*64
    float* e2      = patches + (size_t)CH * B_ * 64;     // CH*B*256
    float* o2      = e2 + (size_t)CH * B_ * 256;         // CH*B*64

    cells_kernel<<<1, 1, 0, stream>>>(focal, cells);

    // x1 = x @ embed1_w^T + embed1_b          (2048 x 4096) x (4096 x 256)
    gemm_bf16_wmma<<<dim3(256 / 64, B_ / 128), 256, 0, stream>>>(
        x, 4096, embed1_w, 4096, embed1_b, x1, 256, 4096);
    // gi1 = x1 @ w_ih1^T + b_ih1
    gemm_bf16_wmma<<<dim3(768 / 64, B_ / 128), 256, 0, stream>>>(
        x1, 256, w_ih1, 256, b_ih1, gi, 768, 256);
    // gh1 = h0 @ w_hh1^T + b_hh1
    gemm_bf16_wmma<<<dim3(768 / 64, B_ / 128), 256, 0, stream>>>(
        h, 256, w_hh1, 256, b_hh1, gh, 768, 256);
    // h1 (written directly into d_out y_h[0])
    gru_ew<<<(B_ * H_) / 256, 256, 0, stream>>>(gi, gh, h, h1, B_, B_);
    // o1 = h1 @ out1_w^T + out1_b
    gemm_bf16_wmma<<<dim3(1, B_ / 128), 256, 0, stream>>>(
        h1, 256, out1_w, 256, out1_b, o1, 64, 256);
    // y = broadcast(o1)
    bcast_y<<<(B_ * 4096) / 256, 256, 0, stream>>>(o1, y);

    if (K > 0) {
        // gh2 = h0 @ w_hh2^T + b_hh2 (shared across all cells)
        gemm_bf16_wmma<<<dim3(768 / 64, B_ / 128), 256, 0, stream>>>(
            h, 256, w_hh2, 256, b_hh2, gh, 768, 256);

        for (int base = 0; base < K; base += CH) {
            int ch = K - base < CH ? K - base : CH;
            int total = ch * B_ * 64;
            gather_patches<<<total / 256, 256, 0, stream>>>(x, cells, base, patches, total);
            // e2 = patches @ embed2_w^T + embed2_b   (ch*2048 x 64) x (64 x 256)
            gemm_bf16_wmma<<<dim3(256 / 64, ch * B_ / 128), 256, 0, stream>>>(
                patches, 64, embed2_w, 64, embed2_b, e2, 256, 64);
            // gi2 = e2 @ w_ih2^T + b_ih2   -- dominant GEMM (~26 GF at K=32)
            gemm_bf16_wmma<<<dim3(768 / 64, ch * B_ / 128), 256, 0, stream>>>(
                e2, 256, w_ih2, 256, b_ih2, gi, 768, 256);
            // h2 chunk written directly into d_out y_h region
            float* h2c = h2 + (size_t)base * B_ * 256;
            gru_ew<<<(ch * B_ * H_) / 256, 256, 0, stream>>>(gi, gh, h, h2c, ch * B_, B_);
            // o2 = h2 @ out2_w^T + out2_b
            gemm_bf16_wmma<<<dim3(1, ch * B_ / 128), 256, 0, stream>>>(
                h2c, 256, out2_w, 256, out2_b, o2, 64, 256);
            scatter_o2<<<total / 256, 256, 0, stream>>>(o2, cells, base, y, total);
        }
    }
}
